// SAModule_16707422781720
// MI455X (gfx1250) — matrix operand order repaired and checked by reading, not compiled, not run on hardware
//
#include <hip/hip_runtime.h>
#include <hip/hip_bf16.h>

// ---------------- problem constants (from reference) ----------------
#define NPTS   16384
#define DIN    64
#define DOUT   128
#define SCTR   4096          // S = N * 0.25
#define KNBR   32
#define NEDGE  (SCTR * KNBR) // 131072
#define DIMF   68            // DIN + 4
#define DPAD   96            // padded feature dim (3 x 32 for WMMA K, 6 x 16 for N)
#define RAD2   0.04f         // RADIUS^2

typedef __attribute__((ext_vector_type(16))) _Float16 v16h;
typedef __attribute__((ext_vector_type(8)))  _Float16 v8h;
typedef __attribute__((ext_vector_type(8)))  float    v8f;

// =====================================================================
// 1) Weight prep: pad to DPAD, transpose to [N][K] so WMMA B-fragment
//    loads are contiguous, convert to f16. Also pad biases.
// =====================================================================
__global__ void prep_weights(const float* __restrict__ W1, const float* __restrict__ b1,
                             const float* __restrict__ W2, const float* __restrict__ b2,
                             const float* __restrict__ W3, const float* __restrict__ b3,
                             _Float16* __restrict__ W1t, _Float16* __restrict__ W2t,
                             _Float16* __restrict__ W3t,
                             float* __restrict__ b1p, float* __restrict__ b2p,
                             float* __restrict__ b3p) {
  int t = blockIdx.x * blockDim.x + threadIdx.x;
  const int w12 = DPAD * DPAD;              // 9216
  const int w3  = DOUT * DPAD;              // 12288
  if (t < w12) {                            // W1t[n][k]
    int n = t / DPAD, k = t % DPAD;
    W1t[t] = (n < DIMF && k < DIMF) ? (_Float16)W1[k * DIMF + n] : (_Float16)0.0f;
  } else if (t < 2 * w12) {                 // W2t[n][k]
    int u = t - w12, n = u / DPAD, k = u % DPAD;
    W2t[u] = (n < DIMF && k < DIMF) ? (_Float16)W2[k * DIMF + n] : (_Float16)0.0f;
  } else if (t < 2 * w12 + w3) {            // W3t[n][k], n<128
    int u = t - 2 * w12, n = u / DPAD, k = u % DPAD;
    W3t[u] = (k < DIMF) ? (_Float16)W3[k * DOUT + n] : (_Float16)0.0f;
  } else if (t < 2 * w12 + w3 + DPAD) {
    int n = t - (2 * w12 + w3);
    b1p[n] = (n < DIMF) ? b1[n] : 0.0f;
  } else if (t < 2 * w12 + w3 + 2 * DPAD) {
    int n = t - (2 * w12 + w3 + DPAD);
    b2p[n] = (n < DIMF) ? b2[n] : 0.0f;
  } else if (t < 2 * w12 + w3 + 2 * DPAD + DOUT) {
    int n = t - (2 * w12 + w3 + 2 * DPAD);
    b3p[n] = b3[n];
  }
}

// =====================================================================
// 2) Farthest point sampling: one persistent 1024-thread block.
//    16 points per thread kept in registers; wave-shuffle argmax.
// =====================================================================
__global__ __launch_bounds__(1024) void fps_kernel(const float* __restrict__ pos,
                                                   int* __restrict__ idx) {
  const int tid = threadIdx.x;
  const int PPT = NPTS / 1024;              // 16
  float px[PPT], py[PPT], pz[PPT], mind[PPT];
  for (int i = 0; i < PPT; ++i) {
    int p = i * 1024 + tid;                 // coalesced
    px[i] = pos[p * 3 + 0];
    py[i] = pos[p * 3 + 1];
    pz[i] = pos[p * 3 + 2];
    mind[i] = 3.4e38f;
  }
  __shared__ float sx, sy, sz;
  __shared__ float swv[32];
  __shared__ int   swi[32];
  __shared__ int   s_next;
  int last = 0;
  for (int s = 0; s < SCTR; ++s) {
    if (tid == 0) {
      idx[s] = last;
      sx = pos[last * 3 + 0]; sy = pos[last * 3 + 1]; sz = pos[last * 3 + 2];
    }
    __syncthreads();
    float bx = sx, by = sy, bz = sz;
    float bestv = -1.0f; int besti = 0;
    #pragma unroll
    for (int i = 0; i < PPT; ++i) {
      float dx = px[i] - bx, dy = py[i] - by, dz = pz[i] - bz;
      float d = dx * dx + dy * dy + dz * dz;
      mind[i] = fminf(mind[i], d);
      if (mind[i] > bestv) { bestv = mind[i]; besti = i * 1024 + tid; }
    }
    // wave32 shuffle reduce (argmax, smallest index on ties)
    #pragma unroll
    for (int off = 16; off > 0; off >>= 1) {
      float ov = __shfl_xor(bestv, off, 32);
      int   oi = __shfl_xor(besti, off, 32);
      if (ov > bestv || (ov == bestv && oi < besti)) { bestv = ov; besti = oi; }
    }
    int wid = tid >> 5;
    if ((tid & 31) == 0) { swv[wid] = bestv; swi[wid] = besti; }
    __syncthreads();
    if (wid == 0) {
      bestv = swv[tid & 31]; besti = swi[tid & 31];
      #pragma unroll
      for (int off = 16; off > 0; off >>= 1) {
        float ov = __shfl_xor(bestv, off, 32);
        int   oi = __shfl_xor(besti, off, 32);
        if (ov > bestv || (ov == bestv && oi < besti)) { bestv = ov; besti = oi; }
      }
      if (tid == 0) s_next = besti;
    }
    __syncthreads();
    last = s_next;
  }
}

// =====================================================================
// 3) Finalize tail outputs: q = pos[idx], batch[idx]=0, idx (int bits).
//    d_out layout: out[S*128] | q[S*3] | batch[S] | idx[S]
// =====================================================================
__global__ void finalize_kernel(const float* __restrict__ pos,
                                const int* __restrict__ idx,
                                float* __restrict__ dout) {
  int s = blockIdx.x * blockDim.x + threadIdx.x;
  if (s >= SCTR) return;
  int id = idx[s];
  float* q = dout + SCTR * DOUT + s * 3;
  q[0] = pos[id * 3 + 0];
  q[1] = pos[id * 3 + 1];
  q[2] = pos[id * 3 + 2];
  int* io = (int*)dout;
  io[SCTR * 131 + s] = 0;   // batch
  io[SCTR * 132 + s] = id;  // idx
}

// =====================================================================
// 4) Radius neighbors: one wave per center, scan in index order using
//    ballot/popcount so the K kept are exactly the K smallest indices.
// =====================================================================
__global__ __launch_bounds__(256) void nbr_kernel(const float* __restrict__ pos,
                                                  const float* __restrict__ dout,
                                                  int* __restrict__ nbr,
                                                  int* __restrict__ cnt) {
  int wave = (blockIdx.x * blockDim.x + threadIdx.x) >> 5;
  int lane = threadIdx.x & 31;
  if (wave >= SCTR) return;
  const float* q = dout + SCTR * DOUT + wave * 3;
  float qx = q[0], qy = q[1], qz = q[2];
  int count = 0;
  for (int base = 0; base < NPTS && count < KNBR; base += 32) {
    int p = base + lane;
    float dx = pos[p * 3 + 0] - qx;
    float dy = pos[p * 3 + 1] - qy;
    float dz = pos[p * 3 + 2] - qz;
    bool in = (dx * dx + dy * dy + dz * dz) <= RAD2;
    unsigned mask = (unsigned)__ballot(in);
    int before = __popc(mask & ((1u << lane) - 1u));
    if (in && (count + before) < KNBR) nbr[wave * KNBR + count + before] = p;
    count += __popc(mask);
  }
  int c = count < KNBR ? count : KNBR;
  if (lane == 0) cnt[wave] = c;
  for (int i = c + lane; i < KNBR; i += 32) nbr[wave * KNBR + i] = 0;
}

// =====================================================================
// 5) Build msg[E, DPAD] in f16: cols 0..63 = x_j, 64..67 = PPF, 68..95 = 0.
//    One thread per (edge, col).
// =====================================================================
__device__ __forceinline__ float angle3(float ax, float ay, float az,
                                        float bx, float by, float bz) {
  float cx = ay * bz - az * by;
  float cy = az * bx - ax * bz;
  float cz = ax * by - ay * bx;
  float cr = sqrtf(cx * cx + cy * cy + cz * cz);
  float dt = ax * bx + ay * by + az * bz;
  return atan2f(cr, dt);
}

__global__ __launch_bounds__(256) void msg_kernel(const float* __restrict__ pos,
                                                  const float* __restrict__ x,
                                                  const float* __restrict__ nrm,
                                                  const float* __restrict__ dout,
                                                  const int* __restrict__ idx,
                                                  const int* __restrict__ nbr,
                                                  const int* __restrict__ cnt,
                                                  _Float16* __restrict__ msg) {
  int gid = blockIdx.x * blockDim.x + threadIdx.x;   // < E*96 = 12.6M, fits int
  int e = gid / DPAD;
  int c = gid - e * DPAD;
  int s = e >> 5, j = e & 31;
  if (c >= DIMF || j >= cnt[s]) { msg[gid] = (_Float16)0.0f; return; }
  int nj = nbr[e];
  if (c < DIN) {
    msg[gid] = (_Float16)x[nj * DIN + c];
    return;
  }
  // PPF scalar (c-64)
  const float* q = dout + SCTR * DOUT + s * 3;
  float psx = pos[nj * 3 + 0] - q[0];
  float psy = pos[nj * 3 + 1] - q[1];
  float psz = pos[nj * 3 + 2] - q[2];
  int ci = idx[s];
  float nix = nrm[ci * 3 + 0], niy = nrm[ci * 3 + 1], niz = nrm[ci * 3 + 2];
  float njx = nrm[nj * 3 + 0], njy = nrm[nj * 3 + 1], njz = nrm[nj * 3 + 2];
  float v;
  int comp = c - DIN;
  if (comp == 0)      v = sqrtf(psx * psx + psy * psy + psz * psz);
  else if (comp == 1) v = angle3(nix, niy, niz, psx, psy, psz);
  else if (comp == 2) v = angle3(njx, njy, njz, psx, psy, psz);
  else                v = angle3(nix, niy, niz, njx, njy, njz);
  msg[gid] = (_Float16)v;
}

// =====================================================================
// 6) WMMA GEMM: Out[M, NT*16] = relu(A[M, KT*32] @ Bt^T + bias).
//    - Whole padded Bt staged into LDS once per block (shared by 8 waves)
//    - All A fragments hoisted to registers; inner loop = ds_load + wmma
//    - f16 output path: wave-private LDS tile transpose so global stores
//      are 6x b128 per lane (coalesced) instead of 48x scalar b16
//    - Pure 32-bit address arithmetic (all offsets < 2^31)
//    A: row-major f16 (lda = KT*32). Bt: [N][K] f16, k-stride = KT*32.
// =====================================================================
template <int NT, int KT, bool OUTF32>
__global__ __launch_bounds__(256) void gemm_wmma_kernel(const _Float16* __restrict__ A,
                                                        const _Float16* __restrict__ Bt,
                                                        const float* __restrict__ bias,
                                                        void* __restrict__ Out,
                                                        int Mtiles, int ldo) {
  constexpr int KW  = KT * 32;        // K width (96)
  constexpr int SBS = KW + 8;         // LDS B row stride in halves (16B aligned)
  constexpr int NR  = NT * 16;        // B rows (N)
  constexpr int OST = NT * 16 + 8;    // LDS out-tile row stride in halves
  __shared__ _Float16 sB[NR * SBS];
  __shared__ _Float16 sO[OUTF32 ? 1 : 8 * 16 * OST];  // 8 wave-private 16xN tiles

  // ---- cooperative stage of Bt into LDS (v8h chunks) ----
  constexpr int CPR = KW / 8;                    // 8-half chunks per row (12)
  for (int ch = threadIdx.x; ch < NR * CPR; ch += 256) {
    int n  = ch / CPR;
    int ko = (ch - n * CPR) * 8;
    *(v8h*)&sB[n * SBS + ko] = *(const v8h*)&Bt[n * KW + ko];
  }
  __syncthreads();

  int wslot = threadIdx.x >> 5;
  int wave = blockIdx.x * 8 + wslot;
  if (wave >= Mtiles) return;                 // wave-uniform: EXEC stays all-ones
  int lane = threadIdx.x & 31;
  int lh = lane >> 4;                         // half-wave select
  int lm = lane & 15;

  // ---- A fragments (16x32 f16 each) hoisted to registers ----
  // lanes 0-15 hold K k0+0..7 / k0+16..23, lanes 16-31 K k0+8..15 / k0+24..31
  const _Float16* arow = A + (wave * 16 + lm) * KW + lh * 8;
  v16h a[KT];
  #pragma unroll
  for (int kt = 0; kt < KT; ++kt) {
    const _Float16* ap = arow + kt * 32;
    v8h alo = *(const v8h*)ap;
    v8h ahi = *(const v8h*)(ap + 16);
    a[kt] = __builtin_shufflevector(alo, ahi, 0, 1, 2, 3, 4, 5, 6, 7,
                                    8, 9, 10, 11, 12, 13, 14, 15);
  }

  _Float16* so = &sO[wslot * 16 * OST];       // this wave's LDS out tile [16][OST]

  // ---- per n-tile: accumulate over K from LDS-resident B ----
  #pragma unroll
  for (int t = 0; t < NT; ++t) {
    v8f acc = {};
    int bbase = (t * 16 + lm) * SBS + lh * 8;
    #pragma unroll
    for (int kt = 0; kt < KT; ++kt) {
      // B fragment (32x16 f16): lanes 0-15 K k0+0..7 / k0+16..23,
      // lanes 16-31 K k0+8..15 / k0+24..31 (two 16B ds loads)
      const _Float16* bp = &sB[bbase + kt * 32];
      v8h blo = *(const v8h*)bp;
      v8h bhi = *(const v8h*)(bp + 16);
      v16h b = __builtin_shufflevector(blo, bhi, 0, 1, 2, 3, 4, 5, 6, 7,
                                       8, 9, 10, 11, 12, 13, 14, 15);
      acc = __builtin_amdgcn_wmma_f32_16x16x32_f16(
          false, a[kt], false, b, (short)0, acc, false, false);
    }
    // epilogue: bias + relu. D layout: vgpr g -> M = g + 8*lh, N = lm.
    int n = t * 16 + lm;
    float bv = bias[n];
    #pragma unroll
    for (int g = 0; g < 8; ++g) {
      int m = g + 8 * lh;
      float v = acc[g] + bv;
      v = v > 0.0f ? v : 0.0f;
      if (OUTF32) {
        ((float*)Out)[(wave * 16 + m) * ldo + n] = v;
      } else {
        so[m * OST + n] = (_Float16)v;        // scatter into wave-private tile
      }
    }
  }

  if (!OUTF32) {
    // Same-wave LDS ops are in-order: no barrier needed; read back our own
    // tile row-contiguously and emit wide coalesced global stores (6x b128).
    constexpr int CPO = (NT * 16) / 8;        // 8-half chunks per out row (12)
    _Float16* op = (_Float16*)Out + wave * 16 * ldo;
    #pragma unroll
    for (int ch = 0; ch < 16 * CPO / 32; ++ch) {
      int u = ch * 32 + lane;
      int m = u / CPO;
      int co = (u - m * CPO) * 8;
      v8h vv = *(const v8h*)&so[m * OST + co];
      *(v8h*)&op[m * ldo + co] = vv;
    }
  }
}

// =====================================================================
// 7) Masked mean aggregation over K edges per center -> agg[S, DPAD] f16
// =====================================================================
__global__ __launch_bounds__(256) void agg_kernel(const _Float16* __restrict__ h2,
                                                  const int* __restrict__ cnt,
                                                  _Float16* __restrict__ agg) {
  int gid = blockIdx.x * blockDim.x + threadIdx.x;
  if (gid >= SCTR * DPAD) return;
  int s = gid / DPAD, c = gid - (gid / DPAD) * DPAD;
  int cn = cnt[s];
  float sum = 0.0f;
  for (int j = 0; j < cn; ++j)
    sum += (float)h2[(s * KNBR + j) * DPAD + c];
  float denom = (float)(cn > 0 ? cn : 1);
  agg[gid] = (_Float16)(sum / denom);
}

// =====================================================================
// Launcher
// =====================================================================
extern "C" void kernel_launch(void* const* d_in, const int* in_sizes, int n_in,
                              void* d_out, int out_size, void* d_ws, size_t ws_size,
                              hipStream_t stream) {
  const float* x   = (const float*)d_in[0];
  const float* pos = (const float*)d_in[1];
  const float* nrm = (const float*)d_in[2];
  // d_in[3] = batch (all zeros, only surfaces as a zero output)
  const float* W1 = (const float*)d_in[4];
  const float* b1 = (const float*)d_in[5];
  const float* W2 = (const float*)d_in[6];
  const float* b2 = (const float*)d_in[7];
  const float* W3 = (const float*)d_in[8];
  const float* b3 = (const float*)d_in[9];
  float* dout = (float*)d_out;

  // ---- workspace carve-up (256B aligned) ----
  size_t off = 0;
  char* ws = (char*)d_ws;
  auto carve = [&](size_t bytes) -> void* {
    void* p = ws + off;
    off += (bytes + 255) & ~(size_t)255;
    return p;
  };
  int*      idxp = (int*)carve((size_t)SCTR * 4);
  int*      nbrp = (int*)carve((size_t)NEDGE * 4);
  int*      cntp = (int*)carve((size_t)SCTR * 4);
  _Float16* msgp = (_Float16*)carve((size_t)NEDGE * DPAD * 2);  // also reused as h2
  _Float16* h1p  = (_Float16*)carve((size_t)NEDGE * DPAD * 2);
  _Float16* aggp = (_Float16*)carve((size_t)SCTR * DPAD * 2);
  _Float16* W1t  = (_Float16*)carve((size_t)DPAD * DPAD * 2);
  _Float16* W2t  = (_Float16*)carve((size_t)DPAD * DPAD * 2);
  _Float16* W3t  = (_Float16*)carve((size_t)DOUT * DPAD * 2);
  float*    b1p  = (float*)carve((size_t)DPAD * 4);
  float*    b2p  = (float*)carve((size_t)DPAD * 4);
  float*    b3p  = (float*)carve((size_t)DOUT * 4);
  (void)ws_size; (void)in_sizes; (void)n_in; (void)out_size;

  // 1) weight prep
  {
    int total = 2 * DPAD * DPAD + DOUT * DPAD + 2 * DPAD + DOUT;
    prep_weights<<<(total + 255) / 256, 256, 0, stream>>>(
        W1, b1, W2, b2, W3, b3, W1t, W2t, W3t, b1p, b2p, b3p);
  }
  // 2) FPS (single persistent block)
  fps_kernel<<<1, 1024, 0, stream>>>(pos, idxp);
  // 3) q / batch / idx outputs
  finalize_kernel<<<(SCTR + 255) / 256, 256, 0, stream>>>(pos, idxp, dout);
  // 4) neighbors (wave per center)
  nbr_kernel<<<(SCTR * 32) / 256, 256, 0, stream>>>(pos, dout, nbrp, cntp);
  // 5) msg build
  {
    int total = NEDGE * DPAD;                 // 12,582,912
    msg_kernel<<<total / 256, 256, 0, stream>>>(
        pos, x, nrm, dout, idxp, nbrp, cntp, msgp);
  }
  // 6) GEMM1: h1 = relu(msg @ W1 + b1)   [E, 96] x [96, 96]
  gemm_wmma_kernel<6, 3, false><<<(NEDGE / 16) / 8, 256, 0, stream>>>(
      msgp, W1t, b1p, h1p, NEDGE / 16, DPAD);
  // 7) GEMM2: h2 = relu(h1 @ W2 + b2)  (h2 reuses msg buffer)
  gemm_wmma_kernel<6, 3, false><<<(NEDGE / 16) / 8, 256, 0, stream>>>(
      h1p, W2t, b2p, msgp, NEDGE / 16, DPAD);
  // 8) masked mean aggregation -> agg[S, 96]
  agg_kernel<<<(SCTR * DPAD) / 256, 256, 0, stream>>>(msgp, cntp, aggp);
  // 9) GEMM3: out = relu(agg @ W3 + b3) -> f32 directly into d_out [S, 128]
  gemm_wmma_kernel<8, 3, true><<<(SCTR / 16) / 8, 256, 0, stream>>>(
      aggp, W3t, b3p, dout, SCTR / 16, DOUT);
}
